// ActorCriticNet_51642686767720
// MI455X (gfx1250) — matrix operand order repaired and checked by reading, not compiled, not run on hardware
//
#include <hip/hip_runtime.h>
#include <hip/hip_bf16.h>

typedef __attribute__((ext_vector_type(16))) _Float16 v16h;
typedef __attribute__((ext_vector_type(8)))  float    v8f;
typedef unsigned int u32x4 __attribute__((ext_vector_type(4)));

#define NPX   2304            // 48*48
#define W1FRAG_DW (7*4*4*32*8)   // 28672 dwords
#define W2FRAG_DW (7*2*32*8)     // 3584 dwords
#define JOBFRAG_DW (4*32*8)      // 1024 dwords
#define FRAG_BYTES ((W1FRAG_DW + W2FRAG_DW + JOBFRAG_DW) * 4)  // 133120

__device__ __forceinline__ float htanh(float x) {
#if __has_builtin(__builtin_amdgcn_tanhf)
    return __builtin_amdgcn_tanhf(x);     // gfx1250 hardware v_tanh_f32 (TRANS op)
#else
    return tanhf(x);
#endif
}

__device__ __forceinline__ unsigned int pack2h(float a, float b) {
    union { _Float16 h[2]; unsigned int u; } p;
    p.h[0] = (_Float16)a; p.h[1] = (_Float16)b;
    return p.u;
}

__device__ __forceinline__ v16h load8dw(const unsigned int* p) {
    union { u32x4 q[2]; v16h h; } u;
    u.q[0] = *reinterpret_cast<const u32x4*>(p);
    u.q[1] = *reinterpret_cast<const u32x4*>(p + 4);
    return u.h;
}

__device__ __forceinline__ v8f wmma_f16(v16h a, v16h b, v8f c) {
    return __builtin_amdgcn_wmma_f32_16x16x32_f16(false, a, false, b, (short)0, c, false, false);
}

// ---------------------------------------------------------------------------
// Prep: pack f32 weights into f16 B-matrix fragments in per-lane VGPR order.
// B packing chosen here, A loads match (16-bit, 32x16): lane%16 = N, lane/16 selects K half (0..15/16..31),
// dword v holds K pair (2v, 2v+1) within the half. 8 consecutive dwords/lane.
// ---------------------------------------------------------------------------
__global__ void prep_frags(const float* __restrict__ w1,   // (7,64,128)
                           const float* __restrict__ w2,   // (7,12,64)
                           const float* __restrict__ jw,   // (6,128)
                           unsigned int* __restrict__ ws) {
    int t = blockIdx.x * 256 + threadIdx.x;
    if (t < W1FRAG_DW) {
        int v = t & 7, lane = (t >> 3) & 31, f = t >> 8;
        int kc = f & 3, nt = (f >> 2) & 3, e = f >> 4;
        int o = nt * 16 + (lane & 15);
        int c = kc * 32 + (lane >> 4) * 16 + v * 2;
        const float* p = w1 + ((size_t)e * 64 + o) * 128 + c;
        ws[t] = pack2h(p[0], p[1]);
    } else if (t < W1FRAG_DW + W2FRAG_DW) {
        int t2 = t - W1FRAG_DW;
        int v = t2 & 7, lane = (t2 >> 3) & 31, f = t2 >> 8;
        int kc = f & 1, e = f >> 1;
        int a = lane & 15;
        int k = kc * 32 + (lane >> 4) * 16 + v * 2;
        float x0 = 0.f, x1 = 0.f;
        if (a < 12) { const float* p = w2 + ((size_t)e * 12 + a) * 64 + k; x0 = p[0]; x1 = p[1]; }
        ws[t] = pack2h(x0, x1);
    } else if (t < W1FRAG_DW + W2FRAG_DW + JOBFRAG_DW) {
        int t3 = t - W1FRAG_DW - W2FRAG_DW;
        int v = t3 & 7, lane = (t3 >> 3) & 31, kc = t3 >> 8;
        int j = lane & 15;
        int c = kc * 32 + (lane >> 4) * 16 + v * 2;
        float x0 = 0.f, x1 = 0.f;
        if (j < 6) { const float* p = jw + (size_t)j * 128 + c; x0 = p[0]; x1 = p[1]; }
        ws[t] = pack2h(x0, x1);
    }
}

// ---------------------------------------------------------------------------
// Main actor kernel: 256 threads = 8 waves, each wave owns 16 pixels.
// grid = 64 images * 18 (= 2304/128) = 1152 workgroups, exact tiling.
// ---------------------------------------------------------------------------
__global__ __launch_bounds__(256, 1) void actor_main(
    const float* __restrict__ x,       // (64,128,2304)
    const float* __restrict__ job_b,   // (6)
    const float* __restrict__ eb1,     // (7,64)
    const float* __restrict__ eb2,     // (7,12)
    const unsigned int* __restrict__ frags,
    float* __restrict__ out)           // (64,12,2304)
{
    __shared__ __align__(16) _Float16 featH[8][16 * 128];  // A staging (f16)
    __shared__ __align__(16) _Float16 Hh[8][16 * 64];      // hidden staging (f16)
    __shared__ float mrS[8][16];
    __shared__ float mfS[8][16];

    const int tid = threadIdx.x;
    const int wv  = tid >> 5;
    const int l   = tid & 31;
    const int ln  = l & 15;    // N-role: output/action/job index; M-role: pixel low part
    const int kh  = l >> 4;    // K half selector / pixel+8 selector

    const int wg = blockIdx.x;
    const int b  = wg / 18;
    const int tp = (wg % 18) * 128 + wv * 16;   // pixel base of this wave's tile

    const float* xb = x + (size_t)b * 128 * NPX;

    // my_robots / my_factories for the 16 pixels
    if (l < 16) {
        float r1 = xb[NPX + tp + l];          // channel 1
        float f0 = xb[tp + l];                // channel 0
        mrS[wv][l] = r1;
        mfS[wv][l] = f0 * (1.0f - r1);
    }

    // Stage feat = x * my_robots into LDS as f16 [px][c].
    // Vectorized: each quad of lanes covers the 16 pixels of one channel via
    // float4 loads; 8 channels per iteration -> 16 iterations total.
    {
        const int p4 = (l & 3) * 4;
        const int cl = l >> 2;            // 0..7
        float m0 = mrS[wv][p4 + 0];
        float m1 = mrS[wv][p4 + 1];
        float m2 = mrS[wv][p4 + 2];
        float m3 = mrS[wv][p4 + 3];
#pragma unroll
        for (int i = 0; i < 16; ++i) {
            int c = i * 8 + cl;
            float4 v4 = *reinterpret_cast<const float4*>(xb + (size_t)c * NPX + tp + p4);
            featH[wv][(p4 + 0) * 128 + c] = (_Float16)(v4.x * m0);
            featH[wv][(p4 + 1) * 128 + c] = (_Float16)(v4.y * m1);
            featH[wv][(p4 + 2) * 128 + c] = (_Float16)(v4.z * m2);
            featH[wv][(p4 + 3) * 128 + c] = (_Float16)(v4.w * m3);
        }
    }

    // Build the 4 A-fragments (M=16 pixels, K=32 channels each) in registers.
    const unsigned int* featU = reinterpret_cast<const unsigned int*>(&featH[wv][0]);
    v16h aF[4];
#pragma unroll
    for (int kc = 0; kc < 4; ++kc) {
        int base = ln * 64 + kc * 16 + kh * 8;
        union { u32x4 q[2]; v16h h; } u;
        u.q[0] = *reinterpret_cast<const u32x4*>(featU + base);
        u.q[1] = *reinterpret_cast<const u32x4*>(featU + base + 4);
        aF[kc] = u.h;
    }

    const unsigned int* w1F  = frags;
    const unsigned int* w2F  = frags + W1FRAG_DW;
    const unsigned int* jobF = frags + W1FRAG_DW + W2FRAG_DW;

    // ---- job head: 16x16x128 via 4 WMMAs, then softmax over j (lanes 0..5) ----
    v8f jacc = {};
#pragma unroll
    for (int kc = 0; kc < 4; ++kc) {
        v16h bf = load8dw(jobF + ((size_t)kc * 32 + l) * 8);
        jacc = wmma_f16(aF[kc], bf, jacc);
    }
    float jb = (ln < 6) ? job_b[ln] : -1e30f;
    float jprob[8];
#pragma unroll
    for (int r = 0; r < 8; ++r) {
        float v = jacc[r] + jb;
        float m = v;
        for (int off = 1; off < 16; off <<= 1) m = fmaxf(m, __shfl_xor(m, off, 32));
        float e = __expf(v - m);
        float s = e;
        for (int off = 1; off < 16; off <<= 1) s += __shfl_xor(s, off, 32);
        jprob[r] = e / s;
    }

    // ---- expert loop: 6 robot experts + 1 factory expert ----
    float accA[8], flA[8];
#pragma unroll
    for (int r = 0; r < 8; ++r) { accA[r] = 0.f; flA[r] = 0.f; }

    for (int e = 0; e < 7; ++e) {
        // stage 1: H = tanh(feat(16x128) @ W1^T(128x64) + b1), 16 WMMAs
#pragma unroll
        for (int nt = 0; nt < 4; ++nt) {
            v8f hacc = {};
#pragma unroll
            for (int kc = 0; kc < 4; ++kc) {
                v16h bf = load8dw(w1F + ((size_t)(((e * 4 + nt) * 4 + kc)) * 32 + l) * 8);
                hacc = wmma_f16(aF[kc], bf, hacc);
            }
            float b1v = eb1[e * 64 + nt * 16 + ln];
#pragma unroll
            for (int r = 0; r < 8; ++r) {
                float t = htanh(hacc[r] + b1v);
                Hh[wv][(r + kh * 8) * 64 + nt * 16 + ln] = (_Float16)t;   // [px][o]
            }
        }

        // stage 2: LG = H(16x64) @ W2^T(64x16pad) + b2, 2 WMMAs
        const unsigned int* HU = reinterpret_cast<const unsigned int*>(&Hh[wv][0]);
        v8f lacc = {};
#pragma unroll
        for (int kc = 0; kc < 2; ++kc) {
            int base = ln * 32 + kc * 16 + kh * 8;
            union { u32x4 q[2]; v16h h; } u;
            u.q[0] = *reinterpret_cast<const u32x4*>(HU + base);
            u.q[1] = *reinterpret_cast<const u32x4*>(HU + base + 4);
            v16h bf = load8dw(w2F + ((size_t)(e * 2 + kc) * 32 + l) * 8);
            lacc = wmma_f16(u.h, bf, lacc);
        }
        float b2v = (ln < 12) ? eb2[e * 12 + ln] : 0.0f;
#pragma unroll
        for (int r = 0; r < 8; ++r) {
            float lg = lacc[r] + b2v;                 // pad lanes contribute 0
            float s = lg;
            for (int off = 1; off < 16; off <<= 1) s += __shfl_xor(s, off, 32);
            float lgn = lg / s;                       // lg / sum over actions
            if (e < 6) accA[r] += lgn * __shfl(jprob[r], e, 16);  // * job_prob[e]
            else       flA[r]  = lgn;
        }
    }

    // ---- blend and store: action_logits = robot*mr + factory*mf ----
    float ov[8];
#pragma unroll
    for (int r = 0; r < 8; ++r) {
        int px = r + kh * 8;
        ov[r] = accA[r] * mrS[wv][px] + flA[r] * mfS[wv][px];
    }
    if (ln < 12) {
        size_t base = ((size_t)b * 12 + ln) * NPX + tp + kh * 8;
        float4 o0 = make_float4(ov[0], ov[1], ov[2], ov[3]);
        float4 o1 = make_float4(ov[4], ov[5], ov[6], ov[7]);
        *reinterpret_cast<float4*>(out + base)     = o0;
        *reinterpret_cast<float4*>(out + base + 4) = o1;
    }
}

// ---------------------------------------------------------------------------
// Value head: small conv chain, scalar f32 (~0.6 GFLOP total)
// ---------------------------------------------------------------------------
__global__ void conv1_k(const float* __restrict__ x, const float* __restrict__ w,
                        const float* __restrict__ bias, float* __restrict__ y) {
    int t = blockIdx.x * 256 + threadIdx.x;
    if (t >= 64 * 32 * 24 * 24) return;
    int ow = t % 24, oh = (t / 24) % 24, oc = (t / 576) % 32, b = t / (576 * 32);
    const float* xb = x + (size_t)b * 128 * NPX;
    const float* wc = w + (size_t)oc * 128 * 4;
    float s = bias[oc];
    int ih = oh * 2, iw = ow * 2;
    for (int ic = 0; ic < 128; ++ic) {
        const float* xp = xb + (size_t)ic * NPX + ih * 48 + iw;
        const float* wp = wc + ic * 4;
        s += xp[0] * wp[0] + xp[1] * wp[1] + xp[48] * wp[2] + xp[49] * wp[3];
    }
    y[t] = htanh(s);
}

__global__ void conv2_k(const float* __restrict__ x, const float* __restrict__ w,
                        const float* __restrict__ bias, float* __restrict__ y) {
    int t = blockIdx.x * 256 + threadIdx.x;
    if (t >= 64 * 16 * 12 * 12) return;
    int ow = t % 12, oh = (t / 12) % 12, oc = (t / 144) % 16, b = t / (144 * 16);
    float s = bias[oc];
    for (int ic = 0; ic < 32; ++ic) {
        const float* xp = x + (((size_t)b * 32 + ic) * 24 + oh * 2) * 24 + ow * 2;
        const float* wp = w + ((size_t)oc * 32 + ic) * 4;
        s += xp[0] * wp[0] + xp[1] * wp[1] + xp[24] * wp[2] + xp[25] * wp[3];
    }
    y[t] = htanh(s);
}

__global__ void conv3_k(const float* __restrict__ x, const float* __restrict__ w,
                        const float* __restrict__ bias, float* __restrict__ y) {
    int t = blockIdx.x * 256 + threadIdx.x;
    if (t >= 64 * 8 * 3 * 3) return;
    int ow = t % 3, oh = (t / 3) % 3, oc = (t / 9) % 8, b = t / 72;
    float s = bias[oc];
    for (int ic = 0; ic < 16; ++ic)
        for (int k2 = 0; k2 < 4; ++k2)
            for (int kw = 0; kw < 4; ++kw) {
                float xv = x[(((size_t)b * 16 + ic) * 12 + oh * 4 + k2) * 12 + ow * 4 + kw];
                float wv = w[(((size_t)oc * 16 + ic) * 4 + k2) * 4 + kw];
                s += xv * wv;
            }
    y[(size_t)b * 72 + oc * 9 + oh * 3 + ow] = htanh(s);   // (B, C*3*3) flatten
}

__global__ void fc_k(const float* __restrict__ v, const float* __restrict__ fw,
                     const float* __restrict__ fb, float* __restrict__ val) {
    int b = threadIdx.x;
    if (b < 64) {
        float s = fb[0];
        for (int k = 0; k < 72; ++k) s += v[(size_t)b * 72 + k] * fw[k];
        val[b] = s;
    }
}

extern "C" void kernel_launch(void* const* d_in, const int* in_sizes, int n_in,
                              void* d_out, int out_size, void* d_ws, size_t ws_size,
                              hipStream_t stream) {
    const float* x    = (const float*)d_in[0];
    const float* jw   = (const float*)d_in[1];
    const float* jb   = (const float*)d_in[2];
    const float* ew1  = (const float*)d_in[3];
    const float* eb1  = (const float*)d_in[4];
    const float* ew2  = (const float*)d_in[5];
    const float* eb2  = (const float*)d_in[6];
    const float* vw1  = (const float*)d_in[7];
    const float* vb1  = (const float*)d_in[8];
    const float* vw2  = (const float*)d_in[9];
    const float* vb2  = (const float*)d_in[10];
    const float* vw3  = (const float*)d_in[11];
    const float* vb3  = (const float*)d_in[12];
    const float* fcw  = (const float*)d_in[13];
    const float* fcb  = (const float*)d_in[14];

    unsigned int* frags = (unsigned int*)d_ws;
    char* wsb = (char*)d_ws;
    float* v1 = (float*)(wsb + FRAG_BYTES);                       // (64,32,24,24)
    float* v2 = (float*)(wsb + FRAG_BYTES + 64*32*24*24*4);       // (64,16,12,12)
    float* v3 = (float*)(wsb + FRAG_BYTES + 64*32*24*24*4 + 64*16*12*12*4); // (64,72)

    float* out = (float*)d_out;
    float* val = out + (size_t)64 * 12 * NPX;

    int prep_n = W1FRAG_DW + W2FRAG_DW + JOBFRAG_DW;
    prep_frags<<<(prep_n + 255) / 256, 256, 0, stream>>>(ew1, ew2, jw, frags);
    actor_main<<<1152, 256, 0, stream>>>(x, jb, eb1, eb2, frags, out);
    conv1_k<<<(64 * 32 * 24 * 24 + 255) / 256, 256, 0, stream>>>(x, vw1, vb1, v1);
    conv2_k<<<(64 * 16 * 12 * 12 + 255) / 256, 256, 0, stream>>>(v1, vw2, vb2, v2);
    conv3_k<<<(64 * 8 * 9 + 255) / 256, 256, 0, stream>>>(v2, vw3, vb3, v3);
    fc_k<<<1, 64, 0, stream>>>(v3, fcw, fcb, val);
}
